// FMoE_10350871183909
// MI455X (gfx1250) — matrix operand order repaired and checked by reading, not compiled, not run on hardware
//
#include <hip/hip_runtime.h>
#include <hip/hip_bf16.h>
#include <math.h>

// ---------------------------------------------------------------------------
// Types for WMMA fragments (CDNA5 gfx1250, wave32)
// ---------------------------------------------------------------------------
typedef __attribute__((ext_vector_type(16))) __bf16 v16bf;
typedef __attribute__((ext_vector_type(8)))  __bf16 v8bf;
typedef __attribute__((ext_vector_type(8)))  float  v8f;

#define EXPERTS 16
#define EHALF   8
#define TOPK    2

// ---------------------------------------------------------------------------
// 1) Gate + route: one wave per token.
//    logits[16] = x @ gate_w + gate_b ; r = x @ route_w + route_b
//    half = (sigmoid(r) > 0.5) ? [0,8) : [8,16); top-2 + softmax over the pair.
// ---------------------------------------------------------------------------
__global__ __launch_bounds__(256) void gate_kernel(
    const float* __restrict__ x, const float* __restrict__ gw,
    const float* __restrict__ gb, const float* __restrict__ rw,
    const float* __restrict__ rb, int* __restrict__ sel_idx,
    float* __restrict__ sel_score, int N, int D)
{
    int wave = (blockIdx.x * blockDim.x + threadIdx.x) >> 5;
    int lane = threadIdx.x & 31;
    if (wave >= N) return;

    const float* xr = x + (size_t)wave * D;
    float acc[17];
#pragma unroll
    for (int i = 0; i < 17; i++) acc[i] = 0.0f;

    for (int d = lane; d < D; d += 32) {
        float xv = xr[d];
        const float* g = gw + (size_t)d * EXPERTS;
#pragma unroll
        for (int e = 0; e < EXPERTS; e++) acc[e] += xv * g[e];
        acc[16] += xv * rw[d];
    }
#pragma unroll
    for (int i = 0; i < 17; i++)
        for (int off = 16; off; off >>= 1)
            acc[i] += __shfl_xor(acc[i], off, 32);

    if (lane == 0) {
        float r = acc[16] + rb[0];
        int base = (r > 0.0f) ? 0 : EHALF;       // sigmoid(r) > 0.5 <=> r > 0
        float v1 = -3.402823466e38f, v2 = -3.402823466e38f;
        int i1 = 0, i2 = 0;
#pragma unroll
        for (int e = 0; e < EHALF; e++) {
            float v = acc[base + e] + gb[base + e];
            if (v > v1)      { v2 = v1; i2 = i1; v1 = v; i1 = e; }
            else if (v > v2) { v2 = v;  i2 = e; }
        }
        float t  = expf(v2 - v1);                // v1 >= v2, stable softmax
        float s1 = 1.0f / (1.0f + t);
        sel_idx[2 * wave]     = base + i1;
        sel_idx[2 * wave + 1] = base + i2;
        sel_score[2 * wave]     = s1;
        sel_score[2 * wave + 1] = t * s1;
    }
}

// ---------------------------------------------------------------------------
// 2) Ordered per-expert slot assignment (exact reference drop semantics).
//    Single wave walks T slots in order with ballot/popc running counters.
// ---------------------------------------------------------------------------
__global__ __launch_bounds__(32) void order_kernel(
    const int* __restrict__ sel_idx, int* __restrict__ slot_of_t,
    int T, int Cap)
{
    int lane = threadIdx.x & 31;
    unsigned lt_mask = (lane == 31) ? 0x7fffffffu : ((1u << lane) - 1u);
    int cnt[EXPERTS];
#pragma unroll
    for (int e = 0; e < EXPERTS; e++) cnt[e] = 0;

    for (int t0 = 0; t0 < T; t0 += 32) {
        int t = t0 + lane;
        int et = sel_idx[t];
        int my_slot = -1;
#pragma unroll
        for (int e = 0; e < EXPERTS; e++) {
            unsigned m = __builtin_amdgcn_ballot_w32(et == e);
            if (et == e) {
                int pos = cnt[e] + __popc(m & lt_mask);
                my_slot = (pos < Cap) ? (e * Cap + pos) : -1;
            }
            cnt[e] += __popc(m);
        }
        slot_of_t[t] = my_slot;
    }
}

// ---------------------------------------------------------------------------
// 3) Dispatch: copy token rows into buf (bf16), one block per (token,slot).
// ---------------------------------------------------------------------------
__global__ __launch_bounds__(256) void dispatch_kernel(
    const float* __restrict__ x, const int* __restrict__ slot_of_t,
    __bf16* __restrict__ buf, int D)
{
    int t = blockIdx.x;
    int s = slot_of_t[t];
    if (s < 0) return;
    const float* xr = x + (size_t)(t >> 1) * D;   // K == 2
    __bf16* br = buf + (size_t)s * D;
    for (int d = threadIdx.x; d < D; d += 256) br[d] = (__bf16)xr[d];
}

// ---------------------------------------------------------------------------
// 4) Transpose + convert fp32 [R,C] -> bf16 [C,R], per-expert (blockIdx.z).
// ---------------------------------------------------------------------------
__global__ __launch_bounds__(256) void transpose_cvt_kernel(
    const float* __restrict__ src, __bf16* __restrict__ dst, int R, int Cc)
{
    __shared__ float tile[32][33];
    const float* s = src + (size_t)blockIdx.z * R * Cc;
    __bf16*      d = dst + (size_t)blockIdx.z * R * Cc;
    int c0 = blockIdx.x * 32, r0 = blockIdx.y * 32;
    int tx = threadIdx.x, ty = threadIdx.y;  // (32, 8)
#pragma unroll
    for (int i = 0; i < 4; i++)
        tile[ty + 8 * i][tx] = s[(size_t)(r0 + ty + 8 * i) * Cc + c0 + tx];
    __syncthreads();
#pragma unroll
    for (int i = 0; i < 4; i++)
        d[(size_t)(c0 + ty + 8 * i) * R + r0 + tx] = (__bf16)tile[tx][ty + 8 * i];
}

// ---------------------------------------------------------------------------
// 5) bf16 WMMA GEMM, per-expert: Out[M,Nn] = epilogue(A[M,Kd] @ Bt[Nn,Kd]^T + bias)
//    One wave computes a 64x64 tile: 4x4 grid of 16x16 accumulators,
//    v_wmma_f32_16x16x32_bf16, fragments loaded straight from L2.
//    Fragment layouts per CDNA5 ISA 7.12.2:
//      A 16x32 bf16: lane(m=lane&15,hi=lane>>4): elem i -> K = i<8 ? hi*8+i
//                    : 16+hi*8+(i-8)
//      B 32x16 bf16: lane(n=lane&15,hi): elem i -> K same as A (i<8 ? hi*8+i : 16+hi*8+(i-8))
//      D 16x16 f32 : elem r -> (M = r + 8*hi, N = lane&15)
// ---------------------------------------------------------------------------
template <bool RELU>
__global__ __launch_bounds__(32) void gemm_bf16_kernel(
    const __bf16* __restrict__ A,   // [E][M][Kd]
    const __bf16* __restrict__ Bt,  // [E][Nn][Kd]  (B transposed)
    const float*  __restrict__ bias,// [E][Nn]
    __bf16* __restrict__ Out,       // [E][M][Nn]
    int M, int Nn, int Kd)
{
    const int e  = blockIdx.z;
    const int n0 = blockIdx.x * 64;
    const int m0 = blockIdx.y * 64;
    const int lane = threadIdx.x & 31;
    const int ml = lane & 15;
    const int hi = lane >> 4;

    const __bf16* Ae = A  + (size_t)e * M  * Kd;
    const __bf16* Be = Bt + (size_t)e * Nn * Kd;
    const float*  be = bias + (size_t)e * Nn;
    __bf16*       Oe = Out + (size_t)e * M * Nn;

    v8f acc[4][4];
    v8f zero = {};
#pragma unroll
    for (int i = 0; i < 4; i++)
#pragma unroll
        for (int j = 0; j < 4; j++) acc[i][j] = zero;

    for (int kk = 0; kk < Kd; kk += 32) {
        v16bf a[4], b[4];
#pragma unroll
        for (int i = 0; i < 4; i++) {
            const __bf16* ap = Ae + (size_t)(m0 + i * 16 + ml) * Kd + kk;
            v8bf lo = *(const v8bf*)(ap + hi * 8);
            v8bf hh = *(const v8bf*)(ap + 16 + hi * 8);
            a[i] = __builtin_shufflevector(lo, hh, 0, 1, 2, 3, 4, 5, 6, 7,
                                           8, 9, 10, 11, 12, 13, 14, 15);
        }
#pragma unroll
        for (int j = 0; j < 4; j++) {
            const __bf16* bp = Be + (size_t)(n0 + j * 16 + ml) * Kd + kk;
            v8bf lo = *(const v8bf*)(bp + hi * 8);
            v8bf hh = *(const v8bf*)(bp + 16 + hi * 8);
            b[j] = __builtin_shufflevector(lo, hh, 0, 1, 2, 3, 4, 5, 6, 7,
                                           8, 9, 10, 11, 12, 13, 14, 15);
        }
#pragma unroll
        for (int i = 0; i < 4; i++)
#pragma unroll
            for (int j = 0; j < 4; j++)
                acc[i][j] = __builtin_amdgcn_wmma_f32_16x16x32_bf16(
                    false, a[i], false, b[j], (short)0, acc[i][j], false, false);
    }

#pragma unroll
    for (int j = 0; j < 4; j++) {
        int col = n0 + j * 16 + ml;
        float bv = be[col];
#pragma unroll
        for (int i = 0; i < 4; i++) {
#pragma unroll
            for (int r = 0; r < 8; r++) {
                int row = m0 + i * 16 + hi * 8 + r;
                float v = acc[i][j][r] + bv;
                if (RELU) v = (v > 0.0f) ? v : 0.0f;
                Oe[(size_t)row * Nn + col] = (__bf16)v;
            }
        }
    }
}

// ---------------------------------------------------------------------------
// 6) Gather + combine: out[n,d] = sum_k score[n,k] * y[slot(n,k), d]
//    Exactly <=2 contributions per element; fully deterministic, no atomics.
// ---------------------------------------------------------------------------
__global__ __launch_bounds__(256) void gather_kernel(
    const __bf16* __restrict__ y, const int* __restrict__ slot_of_t,
    const float* __restrict__ sel_score, float* __restrict__ out,
    int N, int D)
{
    size_t g = (size_t)blockIdx.x * blockDim.x + threadIdx.x;
    size_t total = (size_t)N * D;
    if (g >= total) return;
    int n = (int)(g / D);
    int d = (int)(g - (size_t)n * D);
    float accv = 0.0f;
    int s0 = slot_of_t[2 * n];
    int s1 = slot_of_t[2 * n + 1];
    if (s0 >= 0) accv += sel_score[2 * n]     * (float)y[(size_t)s0 * D + d];
    if (s1 >= 0) accv += sel_score[2 * n + 1] * (float)y[(size_t)s1 * D + d];
    out[g] = accv;
}

// ---------------------------------------------------------------------------
// Host launcher
// ---------------------------------------------------------------------------
extern "C" void kernel_launch(void* const* d_in, const int* in_sizes, int n_in,
                              void* d_out, int out_size, void* d_ws, size_t ws_size,
                              hipStream_t stream)
{
    (void)n_in; (void)out_size; (void)ws_size;
    const float* x   = (const float*)d_in[0];
    const float* gw  = (const float*)d_in[1];
    const float* gb  = (const float*)d_in[2];
    const float* rw  = (const float*)d_in[3];
    const float* rb  = (const float*)d_in[4];
    const float* W1  = (const float*)d_in[5];
    const float* b1  = (const float*)d_in[6];
    const float* W2  = (const float*)d_in[7];
    const float* b2  = (const float*)d_in[8];
    float* out = (float*)d_out;

    const int E = in_sizes[2];           // 16
    const int D = in_sizes[1] / E;       // 1024
    const int N = in_sizes[0] / D;       // 16384
    const int F = in_sizes[6] / E;       // 1024
    const int T = N * TOPK;              // 32768
    const int Cap = 2 * T / E;           // 4096

    // Workspace carve (~321 MB):
    //   bufy : bf16 [E*Cap*D]  -- dispatch buffer, later reused for y
    //   h    : bf16 [E*Cap*F]
    //   w1t  : bf16 [E*F*D]    (W1 transposed per expert)
    //   w2t  : bf16 [E*D*F]    (W2 transposed per expert)
    //   sel_idx[int T], sel_score[float T], slot_of_t[int T]
    char* p = (char*)d_ws;
    __bf16* bufy = (__bf16*)p;  p += (size_t)E * Cap * D * sizeof(__bf16);
    __bf16* h    = (__bf16*)p;  p += (size_t)E * Cap * F * sizeof(__bf16);
    __bf16* w1t  = (__bf16*)p;  p += (size_t)E * F * D * sizeof(__bf16);
    __bf16* w2t  = (__bf16*)p;  p += (size_t)E * D * F * sizeof(__bf16);
    int*    sel_idx   = (int*)p;   p += (size_t)T * sizeof(int);
    float*  sel_score = (float*)p; p += (size_t)T * sizeof(float);
    int*    slot_of_t = (int*)p;   p += (size_t)T * sizeof(int);

    // 1) gate + route (one wave per token)
    gate_kernel<<<(N * 32 + 255) / 256, 256, 0, stream>>>(
        x, gw, gb, rw, rb, sel_idx, sel_score, N, D);

    // 2) ordered capacity bucketing (single wave, exact reference semantics)
    order_kernel<<<1, 32, 0, stream>>>(sel_idx, slot_of_t, T, Cap);

    // 3) dispatch tokens to expert buffers (bf16)
    dispatch_kernel<<<T, 256, 0, stream>>>(x, slot_of_t, bufy, D);

    // 4) weight transpose + bf16 convert (both weights L2-resident afterwards)
    {
        dim3 blk(32, 8);
        dim3 g1(F / 32, D / 32, E);   // W1 [E][D][F] -> w1t [E][F][D]
        transpose_cvt_kernel<<<g1, blk, 0, stream>>>(W1, w1t, D, F);
        dim3 g2(D / 32, F / 32, E);   // W2 [E][F][D] -> w2t [E][D][F]
        transpose_cvt_kernel<<<g2, blk, 0, stream>>>(W2, w2t, F, D);
    }

    // 5) grouped FFN: h = relu(buf @ W1 + b1); y = h @ W2 + b2 (y overwrites buf)
    {
        dim3 blk(32);
        dim3 g1(F / 64, Cap / 64, E);
        gemm_bf16_kernel<true><<<g1, blk, 0, stream>>>(bufy, w1t, b1, h, Cap, F, D);
        dim3 g2(D / 64, Cap / 64, E);
        gemm_bf16_kernel<false><<<g2, blk, 0, stream>>>(h, w2t, b2, bufy, Cap, D, F);
    }

    // 6) gather + score-weighted combine
    {
        size_t total = (size_t)N * D;
        int blocks = (int)((total + 255) / 256);
        gather_kernel<<<blocks, 256, 0, stream>>>(bufy, slot_of_t, sel_score, out, N, D);
    }
}
